// Block_8211977470460
// MI455X (gfx1250) — compile-verified, matrix operand order checked by reading, not run on hardware
//
#include <hip/hip_runtime.h>
#include <hip/hip_bf16.h>

typedef __attribute__((ext_vector_type(16))) __bf16 bf16x16;
typedef __attribute__((ext_vector_type(8)))  float  f32x8;
typedef __attribute__((ext_vector_type(2)))  float  f32x2;
typedef int v4i __attribute__((vector_size(16)));

#define E_DIM 1024
#define S_LEN 2048
#define NB    4
#define HD    16
#define DD    64
#define FF    4096
#define MROWS (NB * S_LEN)   // 8192

// Detect gfx1250 async global->LDS builtin (device pass only; host pass takes fallback)
#if defined(__AMDGCN__) && defined(__has_builtin)
# if __has_builtin(__builtin_amdgcn_global_load_async_to_lds_b128)
#  define HAVE_ASYNC_LDS 1
# endif
#endif
#ifndef HAVE_ASYNC_LDS
# define HAVE_ASYNC_LDS 0
#endif

#if HAVE_ASYNC_LDS
#define ASYNC_GPTR(p) ((__attribute__((address_space(1))) v4i*)(p))
#define ASYNC_LPTR(p) ((__attribute__((address_space(3))) v4i*)(p))
#endif

// ---- fp32 -> bf16 round-to-nearest-even ----
__device__ __forceinline__ unsigned short f2bf(float f) {
    unsigned int u = __float_as_uint(f);
    u += 0x7FFFu + ((u >> 16) & 1u);
    return (unsigned short)(u >> 16);
}
__device__ __forceinline__ unsigned int pack2(float x, float y) {
    return (unsigned int)f2bf(x) | ((unsigned int)f2bf(y) << 16);
}

// elementwise fp32 -> bf16 (n multiple of 2048)
__global__ __launch_bounds__(256)
void cvt_f32_to_bf16(const float* __restrict__ in, unsigned short* __restrict__ out, int n)
{
    const int i = (blockIdx.x * 256 + threadIdx.x) * 8;
    if (i >= n) return;
    float4 a = *(const float4*)(in + i);
    float4 b = *(const float4*)(in + i + 4);
    uint4 o;
    o.x = pack2(a.x, a.y); o.y = pack2(a.z, a.w);
    o.z = pack2(b.x, b.y); o.w = pack2(b.z, b.w);
    *(uint4*)(out + i) = o;
}

// ============================================================
// NT GEMM on bf16 operands: C[M x Nn] = A[M x K] * B[Nn x K]^T (+bias/relu)
// 256 threads = 8 waves; block tile 128x128; wave tile 32x64 (2x4 accums).
// Double-buffered bf16 LDS tiles (padded stride 40 shorts = 80B);
// async global->LDS (ASYNCcnt) when available, else reg-staged stores.
// ============================================================
template<bool HAS_BIAS, bool RELU>
__global__ __launch_bounds__(256)
void gemm_bf16_nt(const unsigned short* __restrict__ A,
                  const unsigned short* __restrict__ B,
                  const float* __restrict__ bias, float* __restrict__ C,
                  int M, int Nn, int K)
{
    __shared__ __align__(16) unsigned short sA[2][128 * 40];
    __shared__ __align__(16) unsigned short sB[2][128 * 40];

    const int t    = threadIdx.x;
    const int lane = t & 31;
    const int wave = t >> 5;
    const int wm   = wave & 3;    // 0..3 -> 32-row sub-tile
    const int wn   = wave >> 2;   // 0..1 -> 64-col sub-tile
    const int bm   = blockIdx.y * 128;
    const int bn   = blockIdx.x * 128;
    const int l16  = lane & 15;
    const int hf   = lane >> 4;

    f32x8 acc[2][4] = {};

    auto compute = [&](int buf) {
        bf16x16 af[2], bfr[4];
        #pragma unroll
        for (int i = 0; i < 2; ++i)
            af[i] = *(const bf16x16*)&sA[buf][(wm * 32 + i * 16 + l16) * 40 + hf * 16];
        #pragma unroll
        for (int j = 0; j < 4; ++j)
            bfr[j] = *(const bf16x16*)&sB[buf][(wn * 64 + j * 16 + l16) * 40 + hf * 16];
        #pragma unroll
        for (int i = 0; i < 2; ++i)
            #pragma unroll
            for (int j = 0; j < 4; ++j)
                acc[i][j] = __builtin_amdgcn_wmma_f32_16x16x32_bf16(
                    false, af[i], false, bfr[j], (short)0, acc[i][j], false, false);
    };

#if HAVE_ASYNC_LDS
    // 128 rows x 32 cols bf16 per tile = 512 x 16B chunks; 2 chunks/thread
    auto stage_async = [&](int buf, int kk2) {
        #pragma unroll
        for (int c = 0; c < 2; ++c) {
            const int ch  = t + c * 256;
            const int row = ch >> 2, sub = ch & 3;
            __builtin_amdgcn_global_load_async_to_lds_b128(
                ASYNC_GPTR(A + (size_t)(bm + row) * K + kk2 + sub * 8),
                ASYNC_LPTR(&sA[buf][row * 40 + sub * 8]), 0, 0);
            __builtin_amdgcn_global_load_async_to_lds_b128(
                ASYNC_GPTR(B + (size_t)(bn + row) * K + kk2 + sub * 8),
                ASYNC_LPTR(&sB[buf][row * 40 + sub * 8]), 0, 0);
        }
    };
    stage_async(0, 0);
    asm volatile("s_wait_asynccnt 0" ::: "memory");
    __syncthreads();
    int cur = 0;
    for (int kk = 0; kk < K; kk += 32) {
        const bool has_next = (kk + 32) < K;
        if (has_next) stage_async(cur ^ 1, kk + 32);   // overlap with WMMA below
        compute(cur);
        if (has_next) {
            asm volatile("s_wait_asynccnt 0" ::: "memory");
            __syncthreads();
            cur ^= 1;
        }
    }
#else
    uint4 ra[2], rb[2];
    auto stage_load = [&](int kk2) {
        #pragma unroll
        for (int c = 0; c < 2; ++c) {
            const int ch  = t + c * 256;
            const int row = ch >> 2, sub = ch & 3;
            ra[c] = *(const uint4*)(A + (size_t)(bm + row) * K + kk2 + sub * 8);
            rb[c] = *(const uint4*)(B + (size_t)(bn + row) * K + kk2 + sub * 8);
        }
    };
    auto stage_store = [&](int buf) {
        #pragma unroll
        for (int c = 0; c < 2; ++c) {
            const int ch  = t + c * 256;
            const int row = ch >> 2, sub = ch & 3;
            *(uint4*)&sA[buf][row * 40 + sub * 8] = ra[c];
            *(uint4*)&sB[buf][row * 40 + sub * 8] = rb[c];
        }
    };
    stage_load(0);
    stage_store(0);
    __syncthreads();
    int cur = 0;
    for (int kk = 0; kk < K; kk += 32) {
        const bool has_next = (kk + 32) < K;
        if (has_next) stage_load(kk + 32);
        compute(cur);
        if (has_next) {
            stage_store(cur ^ 1);
            __syncthreads();
            cur ^= 1;
        }
    }
#endif

    // epilogue: lane 0-15 N=lane; VGPR r: M=r (lanes<16) / M=r+8 (lanes>=16)
    #pragma unroll
    for (int i = 0; i < 2; ++i) {
        #pragma unroll
        for (int j = 0; j < 4; ++j) {
            const int col = bn + wn * 64 + j * 16 + l16;
            float bv = 0.0f;
            if (HAS_BIAS) bv = bias[col];
            #pragma unroll
            for (int r = 0; r < 8; ++r) {
                const int row = bm + wm * 32 + i * 16 + (hf ? (r + 8) : r);
                float v = acc[i][j][r] + bv;
                if (RELU) v = fmaxf(v, 0.0f);
                C[(size_t)row * Nn + col] = v;
            }
        }
    }
}

// ============================================================
// energy[n,q,k,s] = sum_d Q[n,s,q*64+d] * K[n,s,k*64+d]
// one wave per (n,s); 16 chained V_WMMA_F32_16X16X4_F32 (fp32 exact)
// ============================================================
__global__ __launch_bounds__(256)
void attn_energy(const float* __restrict__ Q, const float* __restrict__ Km,
                 float* __restrict__ energy)
{
    const int wave = threadIdx.x >> 5;
    const int lane = threadIdx.x & 31;
    const int pos  = blockIdx.x * 8 + wave;        // n*2048 + s
    const int n    = pos >> 11;
    const int s    = pos & 2047;
    const float* qb = Q  + (size_t)pos * E_DIM;
    const float* kb = Km + (size_t)pos * E_DIM;
    const int l16  = lane & 15;
    const int koff = (lane >> 4) * 2;

    f32x8 c = {};
    #pragma unroll
    for (int step = 0; step < 16; ++step) {
        const int k0 = step * 4 + koff;
        f32x2 a = *(const f32x2*)(qb + l16 * DD + k0);
        f32x2 b = *(const f32x2*)(kb + l16 * DD + k0);
        c = __builtin_amdgcn_wmma_f32_16x16x4_f32(
                false, a, false, b, (short)0, c, false, false);
    }
    #pragma unroll
    for (int r = 0; r < 8; ++r) {
        const int qrow = (lane >> 4) ? (r + 8) : r;
        energy[(((size_t)n * HD + qrow) * HD + l16) * S_LEN + s] = c[r];
    }
}

// softmax over s (row length 2048), scale 1/sqrt(64)=0.125; one block per (n,q,k)
__global__ __launch_bounds__(256)
void softmax_rows(const float* __restrict__ e, float* __restrict__ att)
{
    __shared__ float red[256];
    const size_t base = (size_t)blockIdx.x * S_LEN;
    const int t = threadIdx.x;
    float vals[8];
    float mx = -3.4e38f;
    #pragma unroll
    for (int i = 0; i < 8; ++i) {
        vals[i] = e[base + t + i * 256] * 0.125f;
        mx = fmaxf(mx, vals[i]);
    }
    red[t] = mx; __syncthreads();
    for (int off = 128; off; off >>= 1) {
        if (t < off) red[t] = fmaxf(red[t], red[t + off]);
        __syncthreads();
    }
    mx = red[0]; __syncthreads();
    float sm = 0.0f;
    #pragma unroll
    for (int i = 0; i < 8; ++i) { vals[i] = __expf(vals[i] - mx); sm += vals[i]; }
    red[t] = sm; __syncthreads();
    for (int off = 128; off; off >>= 1) {
        if (t < off) red[t] += red[t + off];
        __syncthreads();
    }
    const float inv = 1.0f / red[0];
    #pragma unroll
    for (int i = 0; i < 8; ++i) att[base + t + i * 256] = vals[i] * inv;
}

// rs[n,q,s] = sum_k att[n,q,k,s]
__global__ __launch_bounds__(256)
void att_rowsum(const float* __restrict__ att, float* __restrict__ rs)
{
    const int idx = blockIdx.x * 256 + threadIdx.x;   // over 4*16*2048
    const int s  = idx & 2047;
    const int nq = idx >> 11;
    float sum = 0.0f;
    #pragma unroll
    for (int k = 0; k < HD; ++k)
        sum += att[((size_t)nq * HD + k) * S_LEN + s];
    rs[idx] = sum;
}

// vhs[n,s,d] = sum_h V[n,s,h*64+d]
__global__ __launch_bounds__(256)
void v_headsum(const float* __restrict__ V, float* __restrict__ vhs)
{
    const int idx = blockIdx.x * 256 + threadIdx.x;   // over 8192*64
    const int d   = idx & 63;
    const int pos = idx >> 6;
    const float* vb = V + (size_t)pos * E_DIM + d;
    float sum = 0.0f;
    #pragma unroll
    for (int h = 0; h < HD; ++h) sum += vb[h * DD];
    vhs[idx] = sum;
}

// out[n,q,s,d] = rs[n,q,s] * vhs[n,s,d]; (H,S,D)->(S,E) reshape is identity on flat idx
__global__ __launch_bounds__(256)
void attn_combine(const float* __restrict__ rs, const float* __restrict__ vhs,
                  float* __restrict__ O)
{
    const int idx = blockIdx.x * 256 + threadIdx.x;   // over N*H*S*D = 8388608
    const int d  = idx & 63;
    const int s  = (idx >> 6) & 2047;
    const int qh = (idx >> 17) & 15;
    const int n  = idx >> 21;
    O[idx] = rs[((size_t)n * HD + qh) * S_LEN + s] *
             vhs[((size_t)n * S_LEN + s) * DD + d];
}

// out = LayerNorm(a + b) * w + bias ; row length 1024, one block per row
__global__ __launch_bounds__(256)
void add_layernorm(const float* __restrict__ a, const float* __restrict__ b,
                   const float* __restrict__ w, const float* __restrict__ bb,
                   float* __restrict__ out)
{
    __shared__ float rsum[256];
    __shared__ float rsq[256];
    const size_t base = (size_t)blockIdx.x * E_DIM;
    const int t = threadIdx.x;
    float v[4]; float s = 0.0f, sq = 0.0f;
    #pragma unroll
    for (int i = 0; i < 4; ++i) {
        v[i] = a[base + t + i * 256] + b[base + t + i * 256];
        s += v[i]; sq += v[i] * v[i];
    }
    rsum[t] = s; rsq[t] = sq; __syncthreads();
    for (int off = 128; off; off >>= 1) {
        if (t < off) { rsum[t] += rsum[t + off]; rsq[t] += rsq[t + off]; }
        __syncthreads();
    }
    const float mean = rsum[0] * (1.0f / (float)E_DIM);
    const float var  = rsq[0] * (1.0f / (float)E_DIM) - mean * mean;
    const float inv  = rsqrtf(var + 1e-5f);
    #pragma unroll
    for (int i = 0; i < 4; ++i) {
        const int c = t + i * 256;
        out[base + c] = (v[i] - mean) * inv * w[c] + bb[c];
    }
}

// ============================================================
extern "C" void kernel_launch(void* const* d_in, const int* in_sizes, int n_in,
                              void* d_out, int out_size, void* d_ws, size_t ws_size,
                              hipStream_t stream)
{
    (void)in_sizes; (void)n_in; (void)out_size; (void)ws_size;
    const float* x    = (const float*)d_in[0];
    const float* Wq   = (const float*)d_in[1];
    const float* Wk   = (const float*)d_in[2];
    const float* Wv   = (const float*)d_in[3];
    const float* Wo   = (const float*)d_in[4];
    const float* bo   = (const float*)d_in[5];
    const float* ln1w = (const float*)d_in[6];
    const float* ln1b = (const float*)d_in[7];
    const float* ln2w = (const float*)d_in[8];
    const float* ln2b = (const float*)d_in[9];
    const float* W1   = (const float*)d_in[10];
    const float* bf1  = (const float*)d_in[11];
    const float* W2   = (const float*)d_in[12];
    const float* bf2  = (const float*)d_in[13];
    float* out = (float*)d_out;

    // ---- workspace carve (fp32 then bf16 regions; all sizes 16B-multiples) ----
    char* p = (char*)d_ws;
    auto allocf = [&](size_t n) { float* r = (float*)p; p += n * sizeof(float); return r; };
    auto alloch = [&](size_t n) { unsigned short* r = (unsigned short*)p; p += n * sizeof(unsigned short); return r; };

    float* Q      = allocf((size_t)MROWS * E_DIM);
    float* Kp     = allocf((size_t)MROWS * E_DIM);
    float* V      = allocf((size_t)MROWS * E_DIM);
    float* energy = allocf((size_t)NB * HD * HD * S_LEN);
    float* att    = allocf((size_t)NB * HD * HD * S_LEN);
    float* rs     = allocf((size_t)NB * HD * S_LEN);
    float* vhs    = allocf((size_t)MROWS * DD);
    float* Oresh  = allocf((size_t)MROWS * E_DIM);
    float* attnp  = allocf((size_t)MROWS * E_DIM);
    float* x1     = allocf((size_t)MROWS * E_DIM);
    float* h1     = allocf((size_t)MROWS * FF);
    float* f2     = allocf((size_t)MROWS * E_DIM);

    unsigned short* xb     = alloch((size_t)MROWS * E_DIM);
    unsigned short* Wqb    = alloch((size_t)E_DIM * E_DIM);
    unsigned short* Wkb    = alloch((size_t)E_DIM * E_DIM);
    unsigned short* Wvb    = alloch((size_t)E_DIM * E_DIM);
    unsigned short* Wob    = alloch((size_t)E_DIM * E_DIM);
    unsigned short* W1b    = alloch((size_t)FF * E_DIM);
    unsigned short* W2b    = alloch((size_t)E_DIM * FF);
    unsigned short* Oreshb = alloch((size_t)MROWS * E_DIM);
    unsigned short* x1b    = alloch((size_t)MROWS * E_DIM);
    unsigned short* h1b    = alloch((size_t)MROWS * FF);

    dim3 blk(256);
    auto cvt = [&](const float* src, unsigned short* dst, size_t n) {
        cvt_f32_to_bf16<<<dim3((unsigned)(n / 2048)), blk, 0, stream>>>(src, dst, (int)n);
    };

    // one-time bf16 copies of weights + input activations
    cvt(x,  xb,  (size_t)MROWS * E_DIM);
    cvt(Wq, Wqb, (size_t)E_DIM * E_DIM);
    cvt(Wk, Wkb, (size_t)E_DIM * E_DIM);
    cvt(Wv, Wvb, (size_t)E_DIM * E_DIM);
    cvt(Wo, Wob, (size_t)E_DIM * E_DIM);
    cvt(W1, W1b, (size_t)FF * E_DIM);
    cvt(W2, W2b, (size_t)E_DIM * FF);

    dim3 gEE(E_DIM / 128, MROWS / 128);   // (8, 64)
    dim3 gEF(FF    / 128, MROWS / 128);   // (32, 64)

    // Q/K/V projections (bf16 WMMA, fp32 accumulate, async double-buffered LDS)
    gemm_bf16_nt<false, false><<<gEE, blk, 0, stream>>>(xb, Wqb, nullptr, Q,  MROWS, E_DIM, E_DIM);
    gemm_bf16_nt<false, false><<<gEE, blk, 0, stream>>>(xb, Wkb, nullptr, Kp, MROWS, E_DIM, E_DIM);
    gemm_bf16_nt<false, false><<<gEE, blk, 0, stream>>>(xb, Wvb, nullptr, V,  MROWS, E_DIM, E_DIM);

    // per-position 16x16 head-Gram in fp32 WMMA; softmax over S; factorized einsum
    attn_energy <<<dim3(MROWS / 8),              blk, 0, stream>>>(Q, Kp, energy);
    softmax_rows<<<dim3(NB * HD * HD),           blk, 0, stream>>>(energy, att);
    att_rowsum  <<<dim3(NB * HD * S_LEN / 256),  blk, 0, stream>>>(att, rs);
    v_headsum   <<<dim3(MROWS * DD / 256),       blk, 0, stream>>>(V, vhs);
    attn_combine<<<dim3(MROWS * E_DIM / 256),    blk, 0, stream>>>(rs, vhs, Oresh);

    // output projection + residual LN1
    cvt(Oresh, Oreshb, (size_t)MROWS * E_DIM);
    gemm_bf16_nt<true, false><<<gEE, blk, 0, stream>>>(Oreshb, Wob, bo, attnp, MROWS, E_DIM, E_DIM);
    add_layernorm<<<dim3(MROWS), blk, 0, stream>>>(attnp, x, ln1w, ln1b, x1);

    // FFN + residual LN2
    cvt(x1, x1b, (size_t)MROWS * E_DIM);
    gemm_bf16_nt<true, true ><<<gEF, blk, 0, stream>>>(x1b, W1b, bf1, h1, MROWS, FF, E_DIM);
    cvt(h1, h1b, (size_t)MROWS * FF);
    gemm_bf16_nt<true, false><<<gEE, blk, 0, stream>>>(h1b, W2b, bf2, f2, MROWS, E_DIM, FF);
    add_layernorm<<<dim3(MROWS), blk, 0, stream>>>(f2, x1, ln2w, ln2b, out);
}
